// LC_33131377721760
// MI455X (gfx1250) — matrix operand order repaired and checked by reading, not compiled, not run on hardware
//
#include <hip/hip_runtime.h>
#include <hip/hip_bf16.h>

#define NR    1024      // boxes
#define NC    151       // classes
#define OBJD  4096      // fmap dim
#define EMBD  20
#define POSD  128
#define KTAIL 160       // padded tail (20 + 128 + 12 zeros)
#define KTOT  4256      // 4096 + 160 (padded K)
#define KREAL 4244      // real dec_w rows
#define NPAD  160       // padded N (151 -> 160)
#define KCH   32        // K per WMMA (bf16)
#define NKCH  133       // 4256 / 32
#define NMS_TH 0.3f
#define BN_EPS 1e-5f

typedef __attribute__((ext_vector_type(8)))  float        v8f;
typedef __attribute__((ext_vector_type(16))) __bf16       v16bf;
typedef __attribute__((ext_vector_type(4)))  unsigned int v4u;
typedef __attribute__((ext_vector_type(8)))  int          v8i;
typedef __attribute__((ext_vector_type(4)))  int          v4i;

union BfFrag { v16bf v; unsigned short u[16]; };

__device__ __forceinline__ unsigned short f2bf(float f) {
  unsigned u = __float_as_uint(f);
  u += 0x7FFFu + ((u >> 16) & 1u);   // round-to-nearest-even
  return (unsigned short)(u >> 16);
}

// --- Tensor Data Mover: load a 160x32 bf16 tile of Bt[160][4256] into LDS ---
// D# group0: count=1, lds_addr, global_addr[56:0], type=2
// D# group1: data_size=1(2B), tensor_dim0=4256, tensor_dim1=160,
//            tile_dim0=32 (64B, mult of 4B), tile_dim1=160, dim0_stride=4256
__device__ __forceinline__ void tdm_load_b_tile(const unsigned short* gsrc, unsigned lds_off) {
  unsigned long long ga = (unsigned long long)gsrc;
  v4u g0;
  g0[0] = 1u;                                     // count=1 (valid descriptor)
  g0[1] = lds_off;                                // LDS byte address
  g0[2] = (unsigned)(ga & 0xFFFFFFFFull);         // global_addr[31:0]
  g0[3] = (unsigned)((ga >> 32) & 0x01FFFFFFull)  // global_addr[56:32]
          | 0x80000000u;                          // type=2 ("image")
  v8i g1;
  g1[0] = 0x00010000;          // data_size=1 -> 2-byte elements
  g1[1] = (int)(4256u << 16);  // tensor_dim0[15:0]=4256 (K length, contiguous)
  g1[2] = (int)(160u  << 16);  // tensor_dim1[15:0]=160 (N rows)
  g1[3] = (int)(32u   << 16);  // tile_dim0=32 (K chunk)
  g1[4] = 160;                 // tile_dim1=160 (all N rows)
  g1[5] = 4256;                // tensor_dim0_stride low32 (elements)
  g1[6] = 0;
  g1[7] = 0;
  v4i gz; gz[0]=gz[1]=gz[2]=gz[3]=0;
#if __clang_major__ >= 23
  v8i gz8; gz8[0]=gz8[1]=gz8[2]=gz8[3]=gz8[4]=gz8[5]=gz8[6]=gz8[7]=0;
  __builtin_amdgcn_tensor_load_to_lds(g0, g1, gz, gz, gz8, 0);
#else
  __builtin_amdgcn_tensor_load_to_lds(g0, g1, gz, gz, 0);
#endif
}

// -------- prep: Bt[n][k] = bf16(dec_w[k][n]) (padded, transposed) -----------
__global__ void prep_bt_kernel(const float* __restrict__ dec_w,
                               unsigned short* __restrict__ Bt) {
  int idx = blockIdx.x * 256 + threadIdx.x;          // < 160*4256
  int n = idx / KTOT;
  int k = idx - n * KTOT;
  float v = (n < NC && k < KREAL) ? dec_w[(size_t)k * NC + n] : 0.0f;
  Bt[idx] = f2bf(v);
}

// ------- prep: tail[row][0:20]=softmax(logits)@embW, [20:148]=pos, pad 0 ----
__global__ void prep_tail_kernel(const float* __restrict__ logits,
                                 const float* __restrict__ embW,
                                 const float* __restrict__ priors,
                                 const float* __restrict__ bng, const float* __restrict__ bnb,
                                 const float* __restrict__ bnm, const float* __restrict__ bnv,
                                 const float* __restrict__ posw, const float* __restrict__ posb,
                                 float* __restrict__ tail) {
  int row = blockIdx.x, t = threadIdx.x;   // 256 threads
  __shared__ float p[NC];
  __shared__ float red[256];
  float x = (t < NC) ? logits[(size_t)row * NC + t] : -3.4e38f;
  red[t] = x; __syncthreads();
  for (int s = 128; s > 0; s >>= 1) { if (t < s) red[t] = fmaxf(red[t], red[t + s]); __syncthreads(); }
  float m = red[0]; __syncthreads();
  float e = (t < NC) ? __expf(x - m) : 0.0f;
  red[t] = e; __syncthreads();
  for (int s = 128; s > 0; s >>= 1) { if (t < s) red[t] += red[t + s]; __syncthreads(); }
  float inv = 1.0f / red[0]; __syncthreads();
  if (t < NC) p[t] = e * inv;
  __syncthreads();

  if (t < EMBD) {
    float a = 0.0f;
    for (int c = 0; c < NC; ++c) a += p[c] * embW[c * EMBD + t];
    tail[(size_t)row * KTAIL + t] = a;
  } else if (t < EMBD + POSD) {
    int j = t - EMBD;
    float x1 = priors[row * 4 + 0], y1 = priors[row * 4 + 1];
    float x2 = priors[row * 4 + 2], y2 = priors[row * 4 + 3];
    float w = x2 - x1 + 1.0f, h = y2 - y1 + 1.0f;
    float cs[4] = { x1 + 0.5f * w, y1 + 0.5f * h, w, h };
    float a = posb[j];
    for (int k = 0; k < 4; ++k) {
      float xn = (cs[k] - bnm[k]) * rsqrtf(bnv[k] + BN_EPS) * bng[k] + bnb[k];
      a += xn * posw[k * POSD + j];
    }
    tail[(size_t)row * KTAIL + t] = fmaxf(a, 0.0f);
  } else if (t < KTAIL) {
    tail[(size_t)row * KTAIL + t] = 0.0f;
  }
}

// --------------- GEMM: dists = [fmaps | tail] @ Bt^T + dec_b ----------------
// 32 blocks x 64 threads (2 waves); wave computes 16 rows x 160 cols.
// B tiles (already bf16) double-buffered in LDS via TDM; A f32->bf16 in regs.
__global__ void __launch_bounds__(64) gemm_wmma_kernel(
    const float* __restrict__ fmaps,          // 1024 x 4096 f32
    const float* __restrict__ tailA,          // 1024 x 160  f32
    const unsigned short* __restrict__ Bt,    // 160 x 4256  bf16 (n-major)
    const float* __restrict__ dec_b,          // 151
    float* __restrict__ out)                  // 1024 x 151
{
  __shared__ __align__(32) unsigned short ldsB[2][NPAD * KCH];   // 2 x 10 KB
  const int tid  = threadIdx.x;
  const int lane = tid & 31;
  const int wv   = tid >> 5;
  const int m0   = blockIdx.x * 32 + wv * 16;
  const unsigned ldsb0 = (unsigned)(unsigned long long)&ldsB[0][0];
  const unsigned ldsb1 = (unsigned)(unsigned long long)&ldsB[1][0];

  v8f acc[10];
#pragma unroll
  for (int i = 0; i < 10; ++i) acc[i] = v8f{0.f,0.f,0.f,0.f,0.f,0.f,0.f,0.f};

  const int kofs = (lane < 16) ? 0 : 16;   // A lane-group K offset (contiguous, matches B)
  const int ksel = (lane < 16) ? 0 : 16;   // B lane-group K offset (contiguous, matches A)
  const int ncol = lane & 15;
  const int mrow = m0 + (lane & 15);

  tdm_load_b_tile(Bt, ldsb0);                       // prologue: chunk 0
  for (int kk = 0; kk < NKCH; ++kk) {
    const int cur = kk & 1;
    if (kk + 1 < NKCH) {                            // prefetch next B tile
      tdm_load_b_tile(Bt + (size_t)(kk + 1) * KCH, cur ? ldsb0 : ldsb1);
      __builtin_amdgcn_s_wait_tensorcnt(1);         // chunk kk done (in-order TDM)
    } else {
      __builtin_amdgcn_s_wait_tensorcnt(0);
    }
    __syncthreads();

    // ---- A fragment: 16x32 bf16 layout; contiguous run [kofs..+15]
    const int kbase = kk * KCH;
    const float* ap = (kbase < OBJD)
        ? (fmaps + (size_t)mrow * OBJD + kbase)
        : (tailA + (size_t)mrow * KTAIL + (kbase - OBJD));
    if (kk + 1 < NKCH) {
      int nk = kbase + KCH;
      const float* pf = (nk < OBJD) ? (fmaps + (size_t)mrow * OBJD + nk)
                                    : (tailA + (size_t)mrow * KTAIL + (nk - OBJD));
      __builtin_prefetch(pf, 0, 1);                 // global_prefetch_b8
    }
    float4 a0 = *reinterpret_cast<const float4*>(ap + kofs);
    float4 a1 = *reinterpret_cast<const float4*>(ap + kofs + 4);
    float4 a2 = *reinterpret_cast<const float4*>(ap + kofs + 8);
    float4 a3 = *reinterpret_cast<const float4*>(ap + kofs + 12);
    BfFrag af;
    af.u[0]=f2bf(a0.x); af.u[1]=f2bf(a0.y); af.u[2]=f2bf(a0.z); af.u[3]=f2bf(a0.w);
    af.u[4]=f2bf(a1.x); af.u[5]=f2bf(a1.y); af.u[6]=f2bf(a1.z); af.u[7]=f2bf(a1.w);
    af.u[8]=f2bf(a2.x); af.u[9]=f2bf(a2.y); af.u[10]=f2bf(a2.z); af.u[11]=f2bf(a2.w);
    af.u[12]=f2bf(a3.x); af.u[13]=f2bf(a3.y); af.u[14]=f2bf(a3.z); af.u[15]=f2bf(a3.w);

#pragma unroll
    for (int nt = 0; nt < 10; ++nt) {
      // B fragment: lane = column, 16 contiguous bf16 (32B, aligned) from LDS
      const v16bf* bp = reinterpret_cast<const v16bf*>(
          &ldsB[cur][(nt * 16 + ncol) * KCH + ksel]);
      v16bf bv = *bp;
      acc[nt] = __builtin_amdgcn_wmma_f32_16x16x32_bf16(
          false, af.v, false, bv, (short)0, acc[nt], false, false);
    }
    __syncthreads();   // protect buffer cur before next-next TDM overwrite
  }

  // epilogue: C/D layout lane L: M = v + (L>=16 ? 8 : 0), N = L%16
  const int rbase = m0 + ((lane >= 16) ? 8 : 0);
#pragma unroll
  for (int nt = 0; nt < 10; ++nt) {
    int col = nt * 16 + ncol;
    if (col < NC) {
      float bias = dec_b[col];
#pragma unroll
      for (int v = 0; v < 8; ++v)
        out[(size_t)(rbase + v) * NC + col] = acc[nt][v] + bias;
    }
  }
}

// ------------------------------- row softmax --------------------------------
__global__ void softmax_kernel(const float* __restrict__ dists, float* __restrict__ probs) {
  int row = blockIdx.x, t = threadIdx.x;
  __shared__ float red[256];
  float x = (t < NC) ? dists[(size_t)row * NC + t] : -3.4e38f;
  red[t] = x; __syncthreads();
  for (int s = 128; s > 0; s >>= 1) { if (t < s) red[t] = fmaxf(red[t], red[t + s]); __syncthreads(); }
  float m = red[0]; __syncthreads();
  float e = (t < NC) ? __expf(x - m) : 0.0f;
  red[t] = e; __syncthreads();
  for (int s = 128; s > 0; s >>= 1) { if (t < s) red[t] += red[t + s]; __syncthreads(); }
  float inv = 1.0f / red[0];
  if (t < NC) probs[(size_t)row * NC + t] = e * inv;
}

// ------------------------- per-class NMS (one block/class) ------------------
__global__ void __launch_bounds__(1024) nms_kernel(const float* __restrict__ probs,
                                                   const float* __restrict__ boxes,  // (N,C,4)
                                                   float* __restrict__ keepscore) {  // (150,N)
  const int c0 = blockIdx.x;       // 0..149 -> class c0+1
  const int cls = c0 + 1;
  const int t = threadIdx.x;       // 0..1023
  __shared__ float skey[NR];
  __shared__ int   sidx[NR];
  __shared__ float bx1[NR], by1[NR], bx2[NR], by2[NR], bar[NR];
  __shared__ unsigned char sup[NR];

  skey[t] = probs[(size_t)t * NC + cls];
  sidx[t] = t;
  __syncthreads();

  // bitonic sort, descending by score (tie -> smaller original index first)
  for (int k = 2; k <= NR; k <<= 1) {
    for (int j = k >> 1; j > 0; j >>= 1) {
      int ixj = t ^ j;
      if (ixj > t) {
        float s0 = skey[t], s1 = skey[ixj];
        int   i0 = sidx[t], i1 = sidx[ixj];
        bool desc = ((t & k) == 0);
        bool bad = desc ? ((s0 < s1) || (s0 == s1 && i0 > i1))
                        : ((s0 > s1) || (s0 == s1 && i0 < i1));
        if (bad) { skey[t] = s1; sidx[t] = i1; skey[ixj] = s0; sidx[ixj] = i0; }
      }
      __syncthreads();
    }
  }

  int oi = sidx[t];
  float x1 = boxes[((size_t)oi * NC + cls) * 4 + 0];
  float y1 = boxes[((size_t)oi * NC + cls) * 4 + 1];
  float x2 = boxes[((size_t)oi * NC + cls) * 4 + 2];
  float y2 = boxes[((size_t)oi * NC + cls) * 4 + 3];
  bx1[t] = x1; by1[t] = y1; bx2[t] = x2; by2[t] = y2;
  bar[t] = (x2 - x1 + 1.0f) * (y2 - y1 + 1.0f);
  sup[t] = 0;

  // greedy suppression scan (lazy IoU, no NxN matrix)
  for (int i = 0; i < NR - 1; ++i) {
    __syncthreads();                 // make prior updates visible
    bool supi = (sup[i] != 0);
    if (!supi && t > i) {
      float xx1 = fmaxf(bx1[i], bx1[t]);
      float yy1 = fmaxf(by1[i], by1[t]);
      float xx2 = fminf(bx2[i], bx2[t]);
      float yy2 = fminf(by2[i], by2[t]);
      float w = fmaxf(xx2 - xx1 + 1.0f, 0.0f);
      float h = fmaxf(yy2 - yy1 + 1.0f, 0.0f);
      float inter = w * h;
      float iou = inter / (bar[i] + bar[t] - inter);
      if (iou > NMS_TH) sup[t] = 1;
    }
  }
  __syncthreads();
  bool keep = (sup[t] == 0);
  keepscore[(size_t)c0 * NR + oi] = keep ? probs[(size_t)oi * NC + cls] : 0.0f;
}

// ----------------------------- argmax -> preds ------------------------------
__global__ void argmax_kernel(const float* __restrict__ keepscore, float* __restrict__ out) {
  int j = blockIdx.x * 256 + threadIdx.x;
  if (j < NR) {
    float best = -1.0f; int bc = 0;
    for (int c0 = 0; c0 < NC - 1; ++c0) {
      float v = keepscore[(size_t)c0 * NR + j];
      if (v > best) { best = v; bc = c0; }
    }
    out[(size_t)NR * NC + j] = (float)(bc + 1);
  }
}

extern "C" void kernel_launch(void* const* d_in, const int* in_sizes, int n_in,
                              void* d_out, int out_size, void* d_ws, size_t ws_size,
                              hipStream_t stream) {
  const float* fmaps  = (const float*)d_in[0];
  const float* logits = (const float*)d_in[1];
  const float* priors = (const float*)d_in[2];
  const float* boxes  = (const float*)d_in[3];
  const float* embW   = (const float*)d_in[4];
  const float* bng    = (const float*)d_in[5];
  const float* bnb    = (const float*)d_in[6];
  const float* bnm    = (const float*)d_in[7];
  const float* bnv    = (const float*)d_in[8];
  const float* posw   = (const float*)d_in[9];
  const float* posb   = (const float*)d_in[10];
  const float* decw   = (const float*)d_in[11];
  const float* decb   = (const float*)d_in[12];
  float* out = (float*)d_out;

  float* ws = (float*)d_ws;
  unsigned short* Bt = (unsigned short*)ws;       // 160*4256 u16 = 340,480 f32 slots
  float* tailA = ws + (size_t)(NPAD * KTOT) / 2;  // 1024*160 f32
  float* probs = tailA + (size_t)NR * KTAIL;      // 1024*151 f32
  float* keep  = probs + (size_t)NR * NC;         // 150*1024 f32

  prep_bt_kernel  <<<(NPAD * KTOT) / 256, 256, 0, stream>>>(decw, Bt);
  prep_tail_kernel<<<NR, 256, 0, stream>>>(logits, embW, priors, bng, bnb, bnm, bnv,
                                           posw, posb, tailA);
  gemm_wmma_kernel<<<NR / 32, 64, 0, stream>>>(fmaps, tailA, Bt, decb, out);
  softmax_kernel  <<<NR, 256, 0, stream>>>(out, probs);
  nms_kernel      <<<NC - 1, NR, 0, stream>>>(probs, boxes, keep);
  argmax_kernel   <<<(NR + 255) / 256, 256, 0, stream>>>(keep, out);
}
